// GNN_59940563583375
// MI455X (gfx1250) — matrix operand order repaired and checked by reading, not compiled, not run on hardware
//
#include <hip/hip_runtime.h>
#include <hip/hip_bf16.h>

typedef _Float16 v16h __attribute__((ext_vector_type(16)));
typedef float    v8f  __attribute__((ext_vector_type(8)));

#define NN 100000
#define NE 1600000

// ---------------- degree / normalization ----------------
__global__ void fill_ones(float* p, int n) {
    int i = blockIdx.x * blockDim.x + threadIdx.x;
    if (i < n) p[i] = 1.0f;  // self-loop contributes 1 to every degree
}

__global__ void degree_accum(const int* __restrict__ dst, float* deg) {
    int e = blockIdx.x * blockDim.x + threadIdx.x;
    if (e < NE) atomicAdd(&deg[dst[e]], 1.0f);
}

__global__ void rsqrt_inplace(float* p, int n) {
    int i = blockIdx.x * blockDim.x + threadIdx.x;
    if (i < n) p[i] = rsqrtf(p[i]);  // deg >= 1 always
}

// ---------------- GEMM1: h1 = x[100000,10] @ W1[10,64] via WMMA f16 ----------------
// One wave = one 16x16 output tile. blockIdx.x = M tile (6250), wave id = N tile (4).
// Branchless: full 10-float row loaded per lane, K placement picked by v_cndmask.
__global__ __launch_bounds__(128) void gemm1_wmma(const float* __restrict__ x,
                                                  const float* __restrict__ W1,
                                                  float* __restrict__ h1) {
    const int lane = threadIdx.x & 31;
    const int wv   = threadIdx.x >> 5;   // N tile 0..3
    const int half = lane >> 4;          // half-wave
    const int lrow = lane & 15;
    const int mbase = blockIdx.x * 16;
    const int n0 = wv * 16;
    const int n  = n0 + lrow;

    // Load full x row (10 floats, contiguous -> b128+b128 tail handled by compiler)
    const float* xr = x + (mbase + lrow) * 10;
    float xv[10];
#pragma unroll
    for (int k = 0; k < 10; ++k) xv[k] = xr[k];

    // Load the 10 valid W1 rows for this lane's column
    float wv1[10];
#pragma unroll
    for (int k = 0; k < 10; ++k) wv1[k] = W1[k * 64 + n];

    // A layout (16-bit A 16x32): half0 elems 0..7 -> K=0..7, elems 8..15 -> K=16..23 (zero, K>=10)
    //                            half1 elems 0..7 -> K=8..15 (only 8,9 valid), 8..15 -> K=24..31 (zero)
    v16h a, b;
#pragma unroll
    for (int e = 0; e < 8; ++e) {
        float av = (half == 0) ? xv[e] : ((e < 2) ? xv[8 + e] : 0.0f);
        a[e] = (_Float16)av;
        a[8 + e] = (_Float16)0.0f;
    }
    // B layout (same as A): half0 elems 0..7 -> K=0..7, half1 elems 0..1 -> K=8,9; all others zero (K>=10)
#pragma unroll
    for (int e = 0; e < 16; ++e) {
        float bv = (e < 8) ? ((half == 0) ? wv1[e & 7] : ((e < 2) ? wv1[8 + (e & 1)] : 0.0f)) : 0.0f;
        b[e] = (_Float16)bv;
    }

    v8f c = {};
    c = __builtin_amdgcn_wmma_f32_16x16x32_f16(false, a, false, b, (short)0, c, false, false);

    const int m0 = mbase + half * 8;
#pragma unroll
    for (int r = 0; r < 8; ++r) h1[(m0 + r) * 64 + n] = c[r];
}

// ---------------- GEMM2: h2 = hr[100000,64] @ W2[64,10] via WMMA f16, K = 2x32 ----------------
__global__ __launch_bounds__(64) void gemm2_wmma(const float* __restrict__ hr,
                                                 const float* __restrict__ W2,
                                                 float* __restrict__ h2) {
    const int lane  = threadIdx.x & 31;
    const int wv    = threadIdx.x >> 5;
    const int half  = lane >> 4;
    const int lrow  = lane & 15;
    const int mbase = (blockIdx.x * 2 + wv) * 16;
    const int ncl   = (lrow < 10) ? lrow : 0;   // clamped column, select below

    v8f c = {};
#pragma unroll
    for (int kc = 0; kc < 2; ++kc) {
        v16h a, b;
#pragma unroll
        for (int e = 0; e < 16; ++e) {
            int kA = kc * 32 + half * 8 + ((e < 8) ? e : (8 + e));
            a[e] = (_Float16)hr[(mbase + lrow) * 64 + kA];
            int kB = kc * 32 + half * 8 + ((e < 8) ? e : (8 + e));
            float bv = W2[kB * 10 + ncl];
            b[e] = (lrow < 10) ? (_Float16)bv : (_Float16)0.0f;
        }
        c = __builtin_amdgcn_wmma_f32_16x16x32_f16(false, a, false, b, (short)0, c, false, false);
    }
    if (lrow < 10) {
        const int m0 = mbase + half * 8;
#pragma unroll
        for (int r = 0; r < 8; ++r) h2[(m0 + r) * 10 + lrow] = c[r];
    }
}

// ---------------- propagation: self loops + edge scatter (float4 data path) ----------------
__global__ void self_scale64(const float4* __restrict__ h4, const float* __restrict__ dinv,
                             float4* __restrict__ out4) {
    int idx = blockIdx.x * blockDim.x + threadIdx.x;   // NN*16 float4s
    if (idx >= NN * 16) return;
    int i = idx >> 4;
    float di = dinv[i];
    float s = di * di;
    float4 v = h4[idx];
    out4[idx] = make_float4(v.x * s, v.y * s, v.z * s, v.w * s);
}

__global__ void scatter64(const int* __restrict__ ei, const float* __restrict__ h,
                          const float* __restrict__ dinv, float* __restrict__ out) {
    int idx = blockIdx.x * blockDim.x + threadIdx.x;   // NE*16 work items (float4 each)
    if (idx >= NE * 16) return;
    int e = idx >> 4, q = (idx & 15) << 2;
    int s = ei[e], d = ei[NE + e];
    float w = dinv[s] * dinv[d];
    const float4 v = *(const float4*)(h + (long long)s * 64 + q);
    float* o = out + (long long)d * 64 + q;
    atomicAdd(o + 0, v.x * w);
    atomicAdd(o + 1, v.y * w);
    atomicAdd(o + 2, v.z * w);
    atomicAdd(o + 3, v.w * w);
}

__global__ void relu_bias64(const float4* __restrict__ agg4, const float* __restrict__ b,
                            float4* __restrict__ hr4) {
    int idx = blockIdx.x * blockDim.x + threadIdx.x;   // NN*16 float4s
    if (idx >= NN * 16) return;
    int c4 = (idx & 15) << 2;
    const float4 bb = *(const float4*)(b + c4);
    float4 v = agg4[idx];
    hr4[idx] = make_float4(fmaxf(v.x + bb.x, 0.0f), fmaxf(v.y + bb.y, 0.0f),
                           fmaxf(v.z + bb.z, 0.0f), fmaxf(v.w + bb.w, 0.0f));
}

__global__ void self_scale10(const float* __restrict__ h, const float* __restrict__ dinv,
                             float* __restrict__ out) {
    int idx = blockIdx.x * blockDim.x + threadIdx.x;
    if (idx >= NN * 10) return;
    int i = idx / 10;
    float di = dinv[i];
    out[idx] = h[idx] * di * di;
}

__global__ void scatter10(const int* __restrict__ ei, const float* __restrict__ h,
                          const float* __restrict__ dinv, float* __restrict__ out) {
    long long idx = (long long)blockIdx.x * blockDim.x + threadIdx.x;
    if (idx >= (long long)NE * 16) return;
    int e = (int)(idx >> 4), cfeat = (int)(idx & 15);
    if (cfeat >= 10) return;
    int s = ei[e], d = ei[NE + e];
    float w = dinv[s] * dinv[d];
    atomicAdd(&out[d * 10 + cfeat], h[s * 10 + cfeat] * w);
}

__global__ void bias_logsoftmax(const float* __restrict__ agg, const float* __restrict__ b2,
                                float* __restrict__ out) {
    int i = blockIdx.x * blockDim.x + threadIdx.x;
    if (i >= NN) return;
    float v[10];
    float m = -3.0e38f;
#pragma unroll
    for (int c = 0; c < 10; ++c) { v[c] = agg[i * 10 + c] + b2[c]; m = fmaxf(m, v[c]); }
    float s = 0.0f;
#pragma unroll
    for (int c = 0; c < 10; ++c) s += expf(v[c] - m);
    float lse = logf(s);
#pragma unroll
    for (int c = 0; c < 10; ++c) out[i * 10 + c] = v[c] - m - lse;
}

extern "C" void kernel_launch(void* const* d_in, const int* in_sizes, int n_in,
                              void* d_out, int out_size, void* d_ws, size_t ws_size,
                              hipStream_t stream) {
    const float* x  = (const float*)d_in[0];
    const int*   ei = (const int*)d_in[1];   // [2, NE] int32 (JAX default x64-off)
    const float* W1 = (const float*)d_in[2];
    const float* b1 = (const float*)d_in[3];
    const float* W2 = (const float*)d_in[4];
    const float* b2 = (const float*)d_in[5];
    float* out = (float*)d_out;

    float* ws   = (float*)d_ws;
    float* dinv = ws;                          // NN
    float* h1   = ws + 102400;                 // NN*64 (16B aligned offset)
    float* a1   = h1 + (size_t)NN * 64;        // NN*64
    float* h2   = a1 + (size_t)NN * 64;        // NN*10
    float* a2   = h2 + (size_t)NN * 10;        // NN*10

    // normalization: deg (with self loops) -> dinv
    fill_ones<<<(NN + 255) / 256, 256, 0, stream>>>(dinv, NN);
    degree_accum<<<(NE + 255) / 256, 256, 0, stream>>>(ei + NE, dinv);
    rsqrt_inplace<<<(NN + 255) / 256, 256, 0, stream>>>(dinv, NN);

    // layer 1
    gemm1_wmma<<<NN / 16, 128, 0, stream>>>(x, W1, h1);
    self_scale64<<<(NN * 16 + 255) / 256, 256, 0, stream>>>((const float4*)h1, dinv, (float4*)a1);
    scatter64<<<(NE * 16 + 255) / 256, 256, 0, stream>>>(ei, h1, dinv, a1);
    relu_bias64<<<(NN * 16 + 255) / 256, 256, 0, stream>>>((const float4*)a1, b1, (float4*)h1);

    // layer 2
    gemm2_wmma<<<NN / 32, 64, 0, stream>>>(h1, W2, h2);
    self_scale10<<<(NN * 10 + 255) / 256, 256, 0, stream>>>(h2, dinv, a2);
    scatter10<<<(int)(((long long)NE * 16 + 255) / 256), 256, 0, stream>>>(ei, h2, dinv, a2);
    bias_logsoftmax<<<(NN + 255) / 256, 256, 0, stream>>>(a2, b2, out);
}
